// SpatialContext_89575837925970
// MI455X (gfx1250) — matrix operand order repaired and checked by reading, not compiled, not run on hardware
//
#include <hip/hip_runtime.h>

// ---------------------------------------------------------------------------
// SpatialContext on MI455X (gfx1250, wave32):
//   k_norm    : per-pixel inverse L2 norm over channels
//   k_costvol : 25-shift cosine-similarity cost volume (LDS halo tiles)
//   k_gemm    : per-pixel 1x1 conv as f16-WMMA GEMM (M=32768,K=1049,N=1024)
//               with f32 accumulate, bias + ReLU epilogue.
//               B fragments are software-pipelined by one tile so the
//               ds_load_b128 -> v_wmma dependency is covered by a full WMMA
//               of execution instead of a raw s_wait_dscnt 0 stall.
// ---------------------------------------------------------------------------

typedef _Float16 half16 __attribute__((ext_vector_type(16)));
typedef _Float16 half8  __attribute__((ext_vector_type(8)));
typedef _Float16 half4v __attribute__((ext_vector_type(4)));
typedef float    f32x8  __attribute__((ext_vector_type(8)));
typedef float    f32x4  __attribute__((ext_vector_type(4)));

#define C_IN  1024
#define HW    4096
#define INC   1049
#define NVOL  25
#define OC    1024
#define BM    128
#define BN    128
#define KSTEPS 33   // 32 feature K-blocks of 32 + one tail block (25 vol + 7 zero)
#define EPSN  1e-12f

// ------------------------------ kernel 1 -----------------------------------
__global__ __launch_bounds__(256) void k_norm(const float* __restrict__ f,
                                              float* __restrict__ inv) {
  int p = blockIdx.x * 256 + threadIdx.x;          // pixel 0..32767
  const float* base = f + (size_t)(p >> 12) * C_IN * HW + (p & (HW - 1));
  float s = 0.f;
#pragma unroll 8
  for (int c = 0; c < C_IN; ++c) {
    float v = base[(size_t)c * HW];
    s = fmaf(v, v, s);
  }
  inv[p] = 1.0f / fmaxf(sqrtf(s), EPSN);
}

// ------------------------------ kernel 2 -----------------------------------
// 16x16 pixel tile per block (+2 halo each side -> 20x20), channels in chunks
// of 16, tile layout [pix][16ch + 4 pad] so float4 channel reads stay 16B
// aligned and bank-spread.
__global__ __launch_bounds__(256) void k_costvol(const float* __restrict__ f,
                                                 const float* __restrict__ inv,
                                                 float* __restrict__ vol) {
  __shared__ __align__(16) float tile[400 * 20];
  __shared__ float invt[400];
  const int t = threadIdx.x;
  const int lx = t & 15, ly = t >> 4;
  const int wx = blockIdx.x, wy = blockIdx.y, b = blockIdx.z;
  const int x0 = wx * 16 - 2, y0 = wy * 16 - 2;

  for (int i = 0; i < 2; ++i) {
    int e = i * 256 + t;
    if (e < 400) {
      int px = e % 20, py = e / 20;
      int gx = x0 + px, gy = y0 + py;
      bool ok = (gx >= 0) & (gx < 64) & (gy >= 0) & (gy < 64);
      invt[e] = ok ? inv[(b << 12) + (gy << 6) + gx] : 0.f;
    }
  }

  float acc[25];
#pragma unroll
  for (int k = 0; k < 25; ++k) acc[k] = 0.f;

  const int pc = (ly + 2) * 20 + (lx + 2);
  const float* fb = f + (size_t)b * C_IN * HW;

  for (int cb = 0; cb < C_IN; cb += 16) {
    __syncthreads();
#pragma unroll
    for (int i = 0; i < 25; ++i) {           // 6400 elems / 256 threads
      int e = i * 256 + t;
      int pix = e % 400, c = e / 400;
      int px = pix % 20, py = pix / 20;
      int gx = x0 + px, gy = y0 + py;
      bool ok = (gx >= 0) & (gx < 64) & (gy >= 0) & (gy < 64);
      tile[pix * 20 + c] = ok ? fb[(size_t)(cb + c) * HW + (gy << 6) + gx] : 0.f;
    }
    __syncthreads();
#pragma unroll
    for (int c4 = 0; c4 < 4; ++c4) {
      f32x4 ctr = *(const f32x4*)(tile + pc * 20 + c4 * 4);
#pragma unroll
      for (int dx = 0; dx < 5; ++dx)
#pragma unroll
        for (int dy = 0; dy < 5; ++dy) {
          f32x4 nb = *(const f32x4*)(tile + ((ly + dy) * 20 + (lx + dx)) * 20 + c4 * 4);
          float s = acc[dx * 5 + dy];
          s = fmaf(ctr[0], nb[0], s);
          s = fmaf(ctr[1], nb[1], s);
          s = fmaf(ctr[2], nb[2], s);
          s = fmaf(ctr[3], nb[3], s);
          acc[dx * 5 + dy] = s;
        }
    }
  }

  const float invc = invt[pc];
  const int hw = ((wy * 16 + ly) << 6) + wx * 16 + lx;
  float* vb = vol + (size_t)b * NVOL * HW + hw;
#pragma unroll
  for (int dx = 0; dx < 5; ++dx)
#pragma unroll
    for (int dy = 0; dy < 5; ++dy) {
      int k = dx * 5 + dy;                    // reference order: cc*5 + r
      vb[(size_t)k * HW] = acc[k] * invc * invt[pc + (dy - 2) * 20 + (dx - 2)];
    }
}

// ------------------------------ kernel 3 -----------------------------------
__device__ __forceinline__ void load_block(int kb, const float* pA, const float* pV,
                                           const float* pB, int kg,
                                           float aR[16], float bR[16]) {
  if (kb < 32) {
    const float* a = pA + (size_t)kb * 32 * HW;
#pragma unroll
    for (int i = 0; i < 16; ++i) aR[i] = a[(size_t)(kg * 16 + i) * HW];
    const float* bp = pB + kb * 32;           // pB already includes +kg*16
#pragma unroll
    for (int i = 0; i < 16; ++i) bR[i] = bp[i];
  } else {                                    // tail: 25 vol channels + zeros
#pragma unroll
    for (int i = 0; i < 16; ++i) {
      int kk = kg * 16 + i;
      aR[i] = (kk < NVOL) ? pV[(size_t)kk * HW] : 0.f;
      bR[i] = (kk < NVOL) ? pB[1024 + i] : 0.f;
    }
  }
}

__device__ __forceinline__ half16 load_bfrag(const _Float16* lB, int tn,
                                             int lane, int kh) {
  const int nrow = 16 * tn + (lane & 15);     // lane = N column
  half8 blo = *(const half8*)(lB + nrow * 40 + kh * 8);
  half8 bhi = *(const half8*)(lB + nrow * 40 + 16 + kh * 8);
  return __builtin_shufflevector(blo, bhi, 0, 1, 2, 3, 4, 5, 6, 7,
                                 8, 9, 10, 11, 12, 13, 14, 15);
}

__global__ __launch_bounds__(256) void k_gemm(const float* __restrict__ feat,
                                              const float* __restrict__ w,
                                              const float* __restrict__ bias,
                                              const float* __restrict__ vol,
                                              float* __restrict__ out) {
  // staging LDS (f16 tiles, row stride 40 halves = 80B) overlaid with the
  // 64KB f32 [N=128][M=128 (+4 pad)] epilogue transpose buffer.
  __shared__ __align__(16) unsigned char smem[BN * 132 * 4];
  _Float16* lA = (_Float16*)smem;                   // [128][40]
  _Float16* lB = (_Float16*)(smem + 128 * 40 * 2);  // [128][40]
  float* eC = (float*)smem;

  const int t = threadIdx.x;
  const int lane = t & 31, wv = t >> 5;
  const int nBase = blockIdx.x * BN;
  const int pix0 = blockIdx.y * BM;
  const int b = pix0 >> 12;
  const int hw0 = pix0 & (HW - 1);

  const int sm = t & 127;   // staging row (A: pixel, B: output channel)
  const int kg = t >> 7;    // which 16-channel half of the 32-K block

  const float* pA = feat + (size_t)b * C_IN * HW + hw0 + sm;
  const float* pV = vol + (size_t)b * NVOL * HW + hw0 + sm;
  const float* pB = w + (size_t)(nBase + sm) * INC + kg * 16;

  float aR[16], bR[16];
  load_block(0, pA, pV, pB, kg, aR, bR);

  f32x8 acc[8] = {};

  for (int kb = 0; kb < KSTEPS; ++kb) {
    // commit prefetched registers to LDS with f32->f16 conversion
#pragma unroll
    for (int g = 0; g < 4; ++g) {
      half4v ha = {(_Float16)aR[g * 4 + 0], (_Float16)aR[g * 4 + 1],
                   (_Float16)aR[g * 4 + 2], (_Float16)aR[g * 4 + 3]};
      *(half4v*)(lA + sm * 40 + kg * 16 + g * 4) = ha;
      half4v hb = {(_Float16)bR[g * 4 + 0], (_Float16)bR[g * 4 + 1],
                   (_Float16)bR[g * 4 + 2], (_Float16)bR[g * 4 + 3]};
      *(half4v*)(lB + sm * 40 + kg * 16 + g * 4) = hb;
    }
    __syncthreads();

    if (kb + 1 < KSTEPS) load_block(kb + 1, pA, pV, pB, kg, aR, bR);

    // A fragment: 16x32 f16 layout (lanes<16: K0-7|K16-23, lanes>=16: K8-15|K24-31)
    const int mrow = 16 * wv + (lane & 15);
    const int kh = lane >> 4;
    half8 alo = *(const half8*)(lA + mrow * 40 + kh * 8);
    half8 ahi = *(const half8*)(lA + mrow * 40 + 16 + kh * 8);
    half16 a = __builtin_shufflevector(alo, ahi, 0, 1, 2, 3, 4, 5, 6, 7,
                                       8, 9, 10, 11, 12, 13, 14, 15);

    // software-pipelined B fragments: load tile tn+1 before issuing WMMA tn
    half16 bf = load_bfrag(lB, 0, lane, kh);
#pragma unroll
    for (int tn = 0; tn < 8; ++tn) {
      half16 bnxt = bf;
      if (tn < 7) bnxt = load_bfrag(lB, tn + 1, lane, kh);
      acc[tn] = __builtin_amdgcn_wmma_f32_16x16x32_f16(
          false, a, false, bf, (short)0, acc[tn], false, false);
      bf = bnxt;
    }
    __syncthreads();
  }

  // epilogue: transpose C through LDS, then coalesced bias+ReLU stores
#pragma unroll
  for (int tn = 0; tn < 8; ++tn) {
    int n = 16 * tn + (lane & 15);
    int m = 16 * wv + 8 * (lane >> 4);
#pragma unroll
    for (int i = 0; i < 8; ++i) eC[n * 132 + m + i] = acc[tn][i];
  }
  __syncthreads();

  const int m4 = (t & 31) * 4;
  float* ob = out + ((size_t)b * OC + nBase) * HW + hw0 + m4;
#pragma unroll
  for (int r = 0; r < 16; ++r) {
    int n = (t >> 5) + r * 8;
    f32x4 v = *(const f32x4*)(eC + n * 132 + m4);
    float bv = bias[nBase + n];
#pragma unroll
    for (int j = 0; j < 4; ++j) v[j] = fmaxf(v[j] + bv, 0.f);
    *(f32x4*)(ob + (size_t)n * HW) = v;
  }
}

// ------------------------------ launcher -----------------------------------
extern "C" void kernel_launch(void* const* d_in, const int* in_sizes, int n_in,
                              void* d_out, int out_size, void* d_ws, size_t ws_size,
                              hipStream_t stream) {
  (void)in_sizes; (void)n_in; (void)out_size; (void)ws_size;
  const float* feat = (const float*)d_in[0];  // [8,1024,64,64]
  const float* w    = (const float*)d_in[1];  // [1024,1049]
  const float* bias = (const float*)d_in[2];  // [1024]
  float* out = (float*)d_out;                 // [8,1024,64,64]

  float* inv = (float*)d_ws;                  // 32768 floats
  float* vol = inv + 32768;                   // 8*25*4096 floats

  k_norm<<<128, 256, 0, stream>>>(feat, inv);
  k_costvol<<<dim3(4, 4, 8), 256, 0, stream>>>(feat, inv, vol);
  k_gemm<<<dim3(8, 256), 256, 0, stream>>>(feat, w, bias, vol, out);
}
